// UTPM_82377472738084
// MI455X (gfx1250) — matrix operand order repaired and checked by reading, not compiled, not run on hardware
//
#include <hip/hip_runtime.h>

// ---------------------------------------------------------------------------
// MI455X (gfx1250, wave32) fused implementation.
//
// Sizes: E=100, N=50, T=8, C=4, U=100, L=1000, B=8192, pairs=4950 (pad 4960)
//
// Phase 0 (prep):  pairIJ table, vv-folded W7 pre-swizzled into WMMA
//                  B-fragment layout (bf16, 8 N-tiles incl. zero pad tile),
//                  label_embedding likewise.
// Phase 1 (attn):  per batch row: two attn pools + h5 + h6. x row cached in
//                  LDS; wave32 shuffle reductions; pure VALU (bandwidth bound).
// Phase 2 (wmma):  h7 = cf @ W7v^T via v_wmma_f32_16x16x32_bf16 with A built
//                  on the fly from h5. B fragments double-buffered into LDS
//                  with GLOBAL_LOAD_ASYNC_TO_LDS_B128 (ASYNCcnt), shared by
//                  all 4 waves of the block. Then h8, emitted as A-fragments.
// Phase 3 (wmma):  out = h8 @ label_embedding via WMMA (K padded 100->128).
// ---------------------------------------------------------------------------

#define E_DIM   100
#define N_SEQ   50
#define T_DIM   8
#define B_ROWS  8192
#define U_DIM   100
#define L_DIM   1000
#define NPAIR   4950
#define NPAIR_P 4960          // padded to multiple of 32
#define KC_P    155           // 4960 / 32 K-chunks for phase 2
#define NT_P    7             // ceil(100/16) live N-tiles for phase 2
#define NT_PS   8             // stored N-tiles per chunk (8th = zeros, pad)
#define KC_L    4             // 128 / 32 K-chunks for phase 3 (K=100 padded)
#define NT_L    63            // ceil(1000/16) N-tiles for phase 3
#define MTILES  (B_ROWS / 16) // 512 row tiles
#define BCHUNK  (NT_PS * 32 * 16 * 2)   // 8192 bytes staged per K-chunk

typedef __attribute__((ext_vector_type(16))) __bf16 v16bf;
typedef __attribute__((ext_vector_type(8)))  float  v8f;

union V16U { v16bf v; unsigned short u[16]; };

static __device__ __forceinline__ unsigned short f2bfbits(float f) {
  union { float f; unsigned u; } c; c.f = f;
  unsigned r = c.u + 0x7FFFu + ((c.u >> 16) & 1u);   // round-to-nearest-even
  return (unsigned short)(r >> 16);
}

static __device__ __forceinline__ int pair_off(int i) {
  return (i * (2 * E_DIM - i - 1)) >> 1;             // start index of row i
}

// p -> (i,j), i<j, i-major (matches np.triu_indices(E,k=1))
static __device__ __forceinline__ void pair_ij(int p, int* pi, int* pj) {
  float disc = 39601.f - 8.f * (float)p;             // (2E-1)^2 - 8p
  int i = (int)((199.f - __builtin_sqrtf(disc)) * 0.5f);
  if (i < 0) i = 0;
  if (i > E_DIM - 2) i = E_DIM - 2;
  while (i < E_DIM - 2 && pair_off(i + 1) <= p) ++i;
  while (i > 0 && pair_off(i) > p) --i;
  *pi = i;
  *pj = i + 1 + (p - pair_off(i));
}

// ---------------------------------------------------------------------------
// Phase 0 kernels
// ---------------------------------------------------------------------------
__global__ __launch_bounds__(256) void prep_pairs_kernel(unsigned* pairIJ) {
  int p = blockIdx.x * 256 + threadIdx.x;
  if (p >= NPAIR_P) return;
  if (p >= NPAIR) { pairIJ[p] = 0u; return; }        // pad -> (0,0); B pad = 0
  int i, j;
  pair_ij(p, &i, &j);
  pairIJ[p] = ((unsigned)i << 16) | (unsigned)j;
}

// W7v pack layout (bf16 B-fragments): elem index = ((kc*8+nt)*32+lane)*16+e
// lane holds column t = nt*16 + (lane&15), K = kc*32 + (lane>>4)*16 + e
// nt==7 is a zero tile so each chunk is exactly 8KB (uniform async staging).
__global__ __launch_bounds__(256) void prep_w7pack_kernel(
    const float* __restrict__ W7, const float* __restrict__ v,
    unsigned short* __restrict__ pack) {
  int idx = blockIdx.x * 256 + threadIdx.x;
  if (idx >= KC_P * NT_PS * 32 * 16) return;
  int e = idx & 15, r = idx >> 4;
  int lane = r & 31; r >>= 5;
  int nt = r & (NT_PS - 1), kc = r >> 3;
  int p = kc * 32 + ((lane >> 4) << 4) + e;
  int t = nt * 16 + (lane & 15);
  float val = 0.f;
  if (p < NPAIR && t < U_DIM) {
    int i, j;
    pair_ij(p, &i, &j);
    float vv = 0.f;
    #pragma unroll
    for (int c = 0; c < 4; ++c) vv += v[i * 4 + c] * v[j * 4 + c];
    val = W7[t * NPAIR + p] * vv;
  }
  pack[idx] = f2bfbits(val);
}

// label_embedding pack: elem index = ((kc*63+nt)*32+lane)*16+e
__global__ __launch_bounds__(256) void prep_lpack_kernel(
    const float* __restrict__ L, unsigned short* __restrict__ pack) {
  int idx = blockIdx.x * 256 + threadIdx.x;
  if (idx >= KC_L * NT_L * 32 * 16) return;
  int e = idx & 15, r = idx >> 4;
  int lane = r & 31; r >>= 5;
  int nt = r % NT_L, kc = r / NT_L;
  int K = kc * 32 + ((lane >> 4) << 4) + e;
  int N = nt * 16 + (lane & 15);
  float val = (K < U_DIM && N < L_DIM) ? L[K * L_DIM + N] : 0.f;
  pack[idx] = f2bfbits(val);
}

// ---------------------------------------------------------------------------
// Phase 1: attention pools + h5 + h6.  1 wave = 1 batch row; 4 waves/block.
// Dynamic LDS per wave: x row 5000, e/w bufs 128, qbuf 200, h5 100 -> 5440 f32
// ---------------------------------------------------------------------------
#define ATTN_STRIDE 5440

__global__ __launch_bounds__(128) void attn_pool_kernel(
    const float* __restrict__ x,
    const float* __restrict__ W1, const float* __restrict__ b1,
    const float* __restrict__ W2,
    const float* __restrict__ W3, const float* __restrict__ b3,
    const float* __restrict__ W4,
    const float* __restrict__ W5, const float* __restrict__ b5,
    const float* __restrict__ W6, const float* __restrict__ b6,
    float* __restrict__ h5g, float* __restrict__ h6g) {
  extern __shared__ float lds[];
  const int lane = threadIdx.x & 31, wave = threadIdx.x >> 5;
  const int row = blockIdx.x * 4 + wave;
  if (row >= B_ROWS) return;

  float* xl  = lds + wave * ATTN_STRIDE;  // [50][100]
  float* e1b = xl + 5000;                 // 64
  float* e2b = e1b + 64;                  // 64
  float* qb  = e2b + 64;                  // 200 (q1|q2)
  float* h5s = qb + 200;                  // 100

  // stream x row into LDS with 128-bit loads
  const float4* xr = (const float4*)(x + (size_t)row * (N_SEQ * E_DIM));
  float4* xl4 = (float4*)xl;
  for (int i = lane; i < (N_SEQ * E_DIM) / 4; i += 32) xl4[i] = xr[i];

  // ---- per-position logits for both pools: 4 lanes per t (8 t's) ----
  const int t = lane >> 2, qd = lane & 3;
  const float* w1r = W1 + t * E_DIM;
  const float* w3r = W3 + t * E_DIM;
  const float b1t = b1[t], b3t = b3[t], w2t = W2[t], w4t = W4[t];

  for (int n = 0; n < N_SEQ; ++n) {
    const float* xn = xl + n * E_DIM;
    float a1 = 0.f, a2 = 0.f;
    for (int e = qd; e < E_DIM; e += 4) {
      float xv = xn[e];
      a1 = fmaf(xv, w1r[e], a1);
      a2 = fmaf(xv, w3r[e], a2);
    }
    a1 += __shfl_xor(a1, 1); a1 += __shfl_xor(a1, 2);
    a2 += __shfl_xor(a2, 1); a2 += __shfl_xor(a2, 2);
    float c1 = (a1 + b1t) * w2t;
    float c2 = (a2 + b3t) * w4t;
    c1 += __shfl_xor(c1, 4); c1 += __shfl_xor(c1, 8); c1 += __shfl_xor(c1, 16);
    c2 += __shfl_xor(c2, 4); c2 += __shfl_xor(c2, 8); c2 += __shfl_xor(c2, 16);
    if (lane == 0) { e1b[n] = expf(c1); e2b[n] = expf(c2); }
  }

  // ---- softmax over N=50 (exp-then-softmax, as in reference) ----
  float va1 = e1b[lane];
  float vb1 = (lane < N_SEQ - 32) ? e1b[lane + 32] : -1e30f;
  float va2 = e2b[lane];
  float vb2 = (lane < N_SEQ - 32) ? e2b[lane + 32] : -1e30f;
  float m1 = fmaxf(va1, vb1), m2 = fmaxf(va2, vb2);
  for (int s = 16; s >= 1; s >>= 1) {
    m1 = fmaxf(m1, __shfl_xor(m1, s));
    m2 = fmaxf(m2, __shfl_xor(m2, s));
  }
  float p1a = expf(va1 - m1), p1b = (lane < N_SEQ - 32) ? expf(vb1 - m1) : 0.f;
  float p2a = expf(va2 - m2), p2b = (lane < N_SEQ - 32) ? expf(vb2 - m2) : 0.f;
  float s1 = p1a + p1b, s2 = p2a + p2b;
  for (int s = 16; s >= 1; s >>= 1) {
    s1 += __shfl_xor(s1, s);
    s2 += __shfl_xor(s2, s);
  }
  e1b[lane] = p1a / s1;
  e2b[lane] = p2a / s2;
  if (lane < N_SEQ - 32) { e1b[lane + 32] = p1b / s1; e2b[lane + 32] = p2b / s2; }

  // ---- weighted sums q1,q2 over N ----
  float q1a = 0.f, q1b = 0.f, q1c = 0.f, q1d = 0.f;
  float q2a = 0.f, q2b = 0.f, q2c = 0.f, q2d = 0.f;
  for (int n = 0; n < N_SEQ; ++n) {
    float wa = e1b[n], wb = e2b[n];
    const float* xn = xl + n * E_DIM;
    float x0 = xn[lane], x1 = xn[lane + 32], x2 = xn[lane + 64];
    q1a = fmaf(wa, x0, q1a); q1b = fmaf(wa, x1, q1b); q1c = fmaf(wa, x2, q1c);
    q2a = fmaf(wb, x0, q2a); q2b = fmaf(wb, x1, q2b); q2c = fmaf(wb, x2, q2c);
    if (lane < 4) {
      float x3 = xn[lane + 96];
      q1d = fmaf(wa, x3, q1d);
      q2d = fmaf(wb, x3, q2d);
    }
  }
  qb[lane] = q1a; qb[lane + 32] = q1b; qb[lane + 64] = q1c;
  qb[100 + lane] = q2a; qb[132 + lane] = q2b; qb[164 + lane] = q2c;
  if (lane < 4) { qb[96 + lane] = q1d; qb[196 + lane] = q2d; }

  // ---- h5 = W5 @ [q1;q2] + b5 ----
  for (int o = lane; o < E_DIM; o += 32) {
    float s = b5[o];
    const float* wr = W5 + o * 200;
    #pragma unroll 4
    for (int k = 0; k < 200; ++k) s = fmaf(wr[k], qb[k], s);
    h5s[o] = s;
    h5g[(size_t)row * E_DIM + o] = s;
  }
  // ---- h6 = W6 @ h5 + b6 ----
  for (int o = lane; o < E_DIM; o += 32) {
    float s = b6[o];
    const float* wr = W6 + o * E_DIM;
    #pragma unroll 4
    for (int k = 0; k < E_DIM; ++k) s = fmaf(wr[k], h5s[k], s);
    h6g[(size_t)row * E_DIM + o] = s;
  }
}

// ---------------------------------------------------------------------------
// Phase 2: h7 = cf @ W7v^T via WMMA; then h8, packed as A-fragments.
// Block = 4 waves; each wave owns one 16-row tile (block covers 64 rows).
// B fragments are staged per K-chunk into double-buffered LDS with
// GLOBAL_LOAD_ASYNC_TO_LDS_B128 and shared by all 4 waves: the async DMA of
// chunk kc+1 overlaps the 7 WMMAs of chunk kc; s_wait_asynccnt + barrier
// publishes the buffer.
// LDS: 2*8KB staging + 19840B pairIJ + 4 waves * 4800 f32 tiles = ~113KB.
// ---------------------------------------------------------------------------
__global__ __launch_bounds__(128) void pair_wmma_kernel(
    const float* __restrict__ h5g, const float* __restrict__ h6g,
    const unsigned* __restrict__ pairIJ,
    const unsigned short* __restrict__ w7pack,
    const float* __restrict__ b7,
    const float* __restrict__ W8, const float* __restrict__ b8,
    unsigned short* __restrict__ apack) {
  extern __shared__ char ldsraw[];
  unsigned* ijl = (unsigned*)(ldsraw + 2 * BCHUNK);            // [4960]
  float* fbase  = (float*)(ldsraw + 2 * BCHUNK + NPAIR_P * 4);
  const int lane = threadIdx.x & 31, wave = threadIdx.x >> 5;
  const int tid  = threadIdx.x;
  const int tile = blockIdx.x * 4 + wave;      // 512 tiles of 16 rows
  const int row0 = tile * 16;

  float* h5t = fbase + wave * 4800; // [16][100]
  float* h6t = h5t + 1600;          // [16][100]
  float* sct = h6t + 1600;          // [16][100] h7 scratch

  // issue async staging of K-chunk kcn into LDS buffer buf (8KB, uniform:
  // 128 threads x 4 x b128).  Inline asm per CDNA5 ISA.
  const unsigned lbase0 = (unsigned)(size_t)ldsraw;
  auto stage = [&](int kcn, int buf) {
    const char* src = (const char*)w7pack + (size_t)kcn * BCHUNK;
    unsigned lb = lbase0 + buf * BCHUNK;
    #pragma unroll
    for (int r = 0; r < 4; ++r) {
      int c = tid + r * 128;                       // chunk of 16 bytes
      unsigned laddr = lb + c * 16;
      unsigned long long ga = (unsigned long long)(src + (size_t)c * 16);
      asm volatile("global_load_async_to_lds_b128 %0, %1, off"
                   :: "v"(laddr), "v"(ga) : "memory");
    }
  };

  // per-wave h5/h6 tiles + block-shared pairIJ table
  for (int i = lane; i < 1600; i += 32) {
    h5t[i] = h5g[(size_t)row0 * E_DIM + i];
    h6t[i] = h6g[(size_t)row0 * E_DIM + i];
  }
  for (int i = tid; i < NPAIR_P; i += 128) ijl[i] = pairIJ[i];

  stage(0, 0);
  asm volatile("s_wait_asynccnt 0x0" ::: "memory");
  __syncthreads();

  const int m = lane & 15, kg = lane >> 4;
  const float* h5row = h5t + m * E_DIM;

  v8f acc[NT_P] = {};
  for (int kc = 0; kc < KC_P; ++kc) {
    if (kc + 1 < KC_P) stage(kc + 1, (kc + 1) & 1);

    // Build bf16 A-fragment on the fly: A[m][K] = h5[m][i_K] * h5[m][j_K]
    // A k order used here (contiguous, same as W7v pack): K = (lane/16)*16 + e
    V16U a;
    #pragma unroll
    for (int e = 0; e < 16; ++e) {
      int kl = kg * 16 + e;
      unsigned ij = ijl[kc * 32 + kl];
      float prod = h5row[ij >> 16] * h5row[ij & 0xFFFFu];
      a.u[e] = f2bfbits(prod);
    }
    const v16bf* bv = (const v16bf*)(ldsraw + (kc & 1) * BCHUNK);
    #pragma unroll
    for (int nt = 0; nt < NT_P; ++nt) {
      v16bf bf = bv[nt * 32 + lane];
      acc[nt] = __builtin_amdgcn_wmma_f32_16x16x32_bf16(
          false, a.v, false, bf, (short)0, acc[nt], false, false);
    }

    asm volatile("s_wait_asynccnt 0x0" ::: "memory");
    __syncthreads();      // publish buffer (kc+1)&1; protect buffer kc&1
  }

  // spill h7 (+b7) to LDS.  C layout: M = j + 8*(lane/16), N = lane&15
  #pragma unroll
  for (int nt = 0; nt < NT_P; ++nt) {
    int tcol = nt * 16 + (lane & 15);
    if (tcol < U_DIM) {
      #pragma unroll
      for (int j = 0; j < 8; ++j)
        sct[(j + 8 * kg) * E_DIM + tcol] = acc[nt][j] + b7[tcol];
    }
  }

  // h8 = W8 @ [h6; h7] + b8  (16 rows x 100 outs, lane-strided), into h5t
  for (int i = lane; i < 1600; i += 32) {
    int mm = i / E_DIM, u = i - mm * E_DIM;
    float s = b8[u];
    const float* wr  = W8 + u * 200;
    const float* h6r = h6t + mm * E_DIM;
    const float* h7r = sct + mm * E_DIM;
    #pragma unroll 4
    for (int k = 0; k < E_DIM; ++k)
      s = fmaf(wr[k], h6r[k], fmaf(wr[100 + k], h7r[k], s));
    h5t[i] = s;
  }

  // emit h8 as packed bf16 A-fragments (K padded to 128 with zeros)
  unsigned short* ap = apack + ((size_t)tile * 4) * 512 + lane * 16;
  #pragma unroll
  for (int kc = 0; kc < KC_L; ++kc) {
    V16U o;
    #pragma unroll
    for (int e = 0; e < 16; ++e) {
      int K = kc * 32 + kg * 16 + e;
      float val = (K < U_DIM) ? h5t[m * E_DIM + K] : 0.f;
      o.u[e] = f2bfbits(val);
    }
    *(v16bf*)(ap + kc * 512) = o.v;
  }
}

// ---------------------------------------------------------------------------
// Phase 3: out = h8 @ label_embedding via WMMA.  Block = 4 waves; block owns
// one 16-row tile; waves split the 63 N-tiles.
// ---------------------------------------------------------------------------
__global__ __launch_bounds__(128) void label_wmma_kernel(
    const unsigned short* __restrict__ apack,
    const unsigned short* __restrict__ lpack,
    float* __restrict__ out) {
  const int lane = threadIdx.x & 31, wave = threadIdx.x >> 5;
  const int rt = blockIdx.x;                     // 512 row tiles
  const v16bf* apv = (const v16bf*)apack + (size_t)rt * 4 * 32 + lane;
  const v16bf* lpv = (const v16bf*)lpack;

  v16bf afr[KC_L];
  #pragma unroll
  for (int kc = 0; kc < KC_L; ++kc) afr[kc] = apv[kc * 32];

  const int nt0 = wave * 16;
  const int nt1 = (nt0 + 16 < NT_L) ? nt0 + 16 : NT_L;
  const int rbase = rt * 16 + 8 * (lane >> 4);
  const int ncol = lane & 15;

  for (int nt = nt0; nt < nt1; ++nt) {
    v8f acc = {};
    #pragma unroll
    for (int kc = 0; kc < KC_L; ++kc) {
      v16bf bf = lpv[(kc * NT_L + nt) * 32 + lane];
      acc = __builtin_amdgcn_wmma_f32_16x16x32_bf16(
          false, afr[kc], false, bf, (short)0, acc, false, false);
    }
    int col = nt * 16 + ncol;
    if (col < L_DIM) {
      #pragma unroll
      for (int j = 0; j < 8; ++j)
        out[(size_t)(rbase + j) * L_DIM + col] = acc[j];
    }
  }
}

// ---------------------------------------------------------------------------
extern "C" void kernel_launch(void* const* d_in, const int* in_sizes, int n_in,
                              void* d_out, int out_size, void* d_ws,
                              size_t ws_size, hipStream_t stream) {
  const float* x  = (const float*)d_in[0];
  const float* v  = (const float*)d_in[1];
  const float* W1 = (const float*)d_in[2];
  const float* b1 = (const float*)d_in[3];
  const float* W2 = (const float*)d_in[4];
  const float* W3 = (const float*)d_in[5];
  const float* b3 = (const float*)d_in[6];
  const float* W4 = (const float*)d_in[7];
  const float* W5 = (const float*)d_in[8];
  const float* b5 = (const float*)d_in[9];
  const float* W6 = (const float*)d_in[10];
  const float* b6 = (const float*)d_in[11];
  const float* W7 = (const float*)d_in[12];
  const float* b7 = (const float*)d_in[13];
  const float* W8 = (const float*)d_in[14];
  const float* b8 = (const float*)d_in[15];
  const float* Lm = (const float*)d_in[16];
  float* out = (float*)d_out;

  // workspace carve-up (256B aligned)
  size_t off = 0;
  char* base = (char*)d_ws;
  auto carve = [&](size_t bytes) {
    void* p = base + off;
    off += (bytes + 255) & ~(size_t)255;
    return p;
  };
  float*          h5g    = (float*)carve((size_t)B_ROWS * E_DIM * 4);
  float*          h6g    = (float*)carve((size_t)B_ROWS * E_DIM * 4);
  unsigned*       pairIJ = (unsigned*)carve((size_t)NPAIR_P * 4);
  unsigned short* w7pack = (unsigned short*)carve((size_t)KC_P * BCHUNK);
  unsigned short* lpack  = (unsigned short*)carve((size_t)KC_L * NT_L * 512 * 2);
  unsigned short* apack  = (unsigned short*)carve((size_t)MTILES * 4 * 512 * 2);
  (void)ws_size; (void)in_sizes; (void)n_in; (void)out_size;

  // Phase 0: prep (tiny; L2-resident outputs reused by every block)
  prep_pairs_kernel<<<(NPAIR_P + 255) / 256, 256, 0, stream>>>(pairIJ);
  prep_w7pack_kernel<<<(KC_P * NT_PS * 512 + 255) / 256, 256, 0, stream>>>(
      W7, v, w7pack);
  prep_lpack_kernel<<<(KC_L * NT_L * 512 + 255) / 256, 256, 0, stream>>>(
      Lm, lpack);

  // Phase 1: attention pools + h5 + h6 (bandwidth-bound pass over x)
  attn_pool_kernel<<<B_ROWS / 4, 128, 4 * ATTN_STRIDE * 4, stream>>>(
      x, W1, b1, W2, W3, b3, W4, W5, b5, W6, b6, h5g, h6g);

  // Phase 2: cross-feature GEMM (WMMA + async LDS staging) + h8
  size_t lds2 = 2 * BCHUNK + NPAIR_P * 4 + 4 * 4800 * 4;
  pair_wmma_kernel<<<MTILES / 4, 128, lds2, stream>>>(
      h5g, h6g, pairIJ, w7pack, b7, W8, b8, apack);

  // Phase 3: label-embedding projection (WMMA)
  label_wmma_kernel<<<MTILES, 128, 0, stream>>>(apack, lpack, out);
}
